// MultiHeadSelfAttention_33586644254768
// MI455X (gfx1250) — matrix operand order repaired and checked by reading, not compiled, not run on hardware
//
#include <hip/hip_runtime.h>
#include <hip/hip_bf16.h>

#define DEV static __device__ __forceinline__

typedef _Float16 half_t;
typedef __attribute__((ext_vector_type(16))) _Float16 v16h;
typedef __attribute__((ext_vector_type(8)))  _Float16 v8h;
typedef __attribute__((ext_vector_type(4)))  _Float16 v4h;
typedef __attribute__((ext_vector_type(8)))  float    v8f;
typedef __attribute__((ext_vector_type(4)))  float    v4f;

static constexpr int BATCH = 2;
static constexpr int SEQ   = 4096;
static constexpr int HID   = 768;
static constexpr int HEADS = 12;
static constexpr int HDIM  = 64;
static constexpr int TOK   = BATCH * SEQ;   // 8192

// ---------------------------------------------------------------------------
// WMMA helper: D = A(16x32 f16) * B(32x16 f16) + C(16x16 f32)
// ---------------------------------------------------------------------------
DEV v8f wmma16(v16h a, v16h b, v8f c) {
  return __builtin_amdgcn_wmma_f32_16x16x32_f16(
      /*neg_a=*/false, a, /*neg_b=*/false, b,
      /*c_mod=*/(short)0, c, /*reuse_a=*/false, /*reuse_b=*/false);
}

// A-fragment (16x32, f16). A row-major with leading dim lda.
// lane&15 = row M; VGPR0..3 hold K=base+0..7, VGPR4..7 K=base+8..15,
// base = (lane>>4)*16  ->  two contiguous 16B loads per lane.
DEV v16h load_a(const half_t* A, int lda, int row, int k0, int lane) {
  int r  = row + (lane & 15);
  int kb = k0 + ((lane >> 4) << 4);
  const half_t* p = A + (size_t)r * lda + kb;
  v8h x0 = *(const v8h*)(p);
  v8h x1 = *(const v8h*)(p + 8);
  return __builtin_shufflevector(x0, x1, 0,1,2,3,4,5,6,7,8,9,10,11,12,13,14,15);
}

// B-fragment (32x16, f16) with B[k][n] = W[n][k], W row-major leading dim ldw.
// lane&15 = col N; lanes 0-15 hold K=k0..k0+15, lanes 16-31 K=+16..31
// -> one contiguous 32-byte load per lane.
DEV v16h load_b(const half_t* W, int ldw, int col, int k0, int lane) {
  int n  = col + (lane & 15);
  int kb = k0 + ((lane >> 4) << 4);
  return *(const v16h*)(W + (size_t)n * ldw + kb);
}

// ---------------------------------------------------------------------------
// f32 -> f16 conversion (4 elements/thread)
// ---------------------------------------------------------------------------
__global__ __launch_bounds__(256)
void cvt_f16(const float* __restrict__ in, half_t* __restrict__ out, int n4) {
  int i = blockIdx.x * 256 + threadIdx.x;
  if (i < n4) {
    v4f x = *(const v4f*)(in + (size_t)i * 4);
    v4h y;
    y[0] = (half_t)x[0]; y[1] = (half_t)x[1];
    y[2] = (half_t)x[2]; y[3] = (half_t)x[3];
    *(v4h*)(out + (size_t)i * 4) = y;
  }
}

// ---------------------------------------------------------------------------
// GEMM: out = A[M,K] * W[N,K]^T + bias   (torch Linear)
// 256 threads = 8 waves; wave tile 32x64; block tile 128x128.
// MODE 0: f16 out row-major [M,N]
// MODE 1: f16 out transposed [N,M]   (used for V so attention reads Vt rows)
// MODE 2: f32 out row-major [M,N]    (final output projection)
// ---------------------------------------------------------------------------
template<int MODE>
__global__ __launch_bounds__(256)
void gemm_wmma(const half_t* __restrict__ A, const half_t* __restrict__ W,
               const float* __restrict__ bias, void* __restrict__ outv,
               int M, int N, int K) {
  const int lane = threadIdx.x & 31;
  const int wave = threadIdx.x >> 5;
  const int row0 = blockIdx.y * 128 + (wave & 3) * 32;
  const int col0 = blockIdx.x * 128 + (wave >> 2) * 64;

  v8f acc[2][4] = {};

  for (int k0 = 0; k0 < K; k0 += 32) {
    __builtin_prefetch(A + (size_t)(row0 + (lane & 15)) * K + k0 + 64, 0, 1);
    v16h a0 = load_a(A, K, row0,      k0, lane);
    v16h a1 = load_a(A, K, row0 + 16, k0, lane);
    v16h b[4];
#pragma unroll
    for (int g = 0; g < 4; ++g) b[g] = load_b(W, K, col0 + g * 16, k0, lane);
#pragma unroll
    for (int g = 0; g < 4; ++g) {
      acc[0][g] = wmma16(a0, b[g], acc[0][g]);
      acc[1][g] = wmma16(a1, b[g], acc[1][g]);
    }
  }

  const int lm = lane & 15, lh = lane >> 4;
#pragma unroll
  for (int f = 0; f < 2; ++f) {
#pragma unroll
    for (int g = 0; g < 4; ++g) {
      int n = col0 + g * 16 + lm;
      float bv = bias[n];
      if (MODE == 1) {
        // rows r..r+7 are contiguous per lane -> single 16B store
        v8h pk;
#pragma unroll
        for (int r = 0; r < 8; ++r) pk[r] = (half_t)(acc[f][g][r] + bv);
        int m = row0 + f * 16 + lh * 8;
        *(v8h*)((half_t*)outv + (size_t)n * M + m) = pk;
      } else {
#pragma unroll
        for (int r = 0; r < 8; ++r) {
          int m = row0 + f * 16 + lh * 8 + r;
          float v = acc[f][g][r] + bv;
          if (MODE == 0) ((half_t*)outv)[(size_t)m * N + n] = (half_t)v;
          else           ((float*)outv)[(size_t)m * N + n] = v;
        }
      }
    }
  }
}

// ---------------------------------------------------------------------------
// Flash attention, S^T formulation.
// 128 threads = 4 waves; each wave owns 32 query rows, streams 32-key tiles.
// S^T = K_tile(A, 32x64) x Q^T(B, 64x32): C-layout puts a softmax row (fixed q)
// inside one lane (over VGPRs/frags) + one xor-16 shuffle for the lane halves.
// P^T C-layout gives 8 consecutive keys per lane -> 4 ds_store_b128 into LDS,
// re-read as P A-fragments for O += P x V (Vt rows are contiguous in keys).
// ---------------------------------------------------------------------------
__global__ __launch_bounds__(128)
void flash_attn(const half_t* __restrict__ Q, const half_t* __restrict__ Kmat,
                const half_t* __restrict__ Vt, half_t* __restrict__ O) {
  __shared__ __align__(16) half_t P[4][32 * 32];
  const int lane = threadIdx.x & 31;
  const int wave = threadIdx.x >> 5;
  const int lm = lane & 15, lh = lane >> 4;
  const int q0 = blockIdx.x * 128 + wave * 32;
  const int h  = blockIdx.y;
  const int b  = blockIdx.z;

  const half_t* Qp = Q    + (size_t)(b * SEQ) * HID + h * HDIM;
  const half_t* Kp = Kmat + (size_t)(b * SEQ) * HID + h * HDIM;
  const half_t* Vp = Vt   + (size_t)(h * HDIM) * TOK + b * SEQ;
  half_t* pw = P[wave];

  // Q as B-operand fragments, resident all loop; fold in 1/sqrt(64) (exact pow2)
  v16h bq[2][2];
#pragma unroll
  for (int qf = 0; qf < 2; ++qf)
#pragma unroll
    for (int ks = 0; ks < 2; ++ks) {
      v16h t = load_b(Qp, HID, q0 + qf * 16, ks * 32, lane);
#pragma unroll
      for (int i = 0; i < 16; ++i) t[i] = t[i] * (_Float16)0.125f;
      bq[qf][ks] = t;
    }

  v8f o[2][4] = {};
  float rm[2] = { -3.0e38f, -3.0e38f };
  float rs[2] = { 0.0f, 0.0f };

  for (int kv = 0; kv < SEQ; kv += 32) {
    // S^T tile: st[kf][qf]; element (key = kf*16 + r + 8*lh, q = qf*16 + lm)
    v8f st[2][2] = {};
#pragma unroll
    for (int ks = 0; ks < 2; ++ks) {
      v16h ak0 = load_a(Kp, HID, kv,      ks * 32, lane);
      v16h ak1 = load_a(Kp, HID, kv + 16, ks * 32, lane);
#pragma unroll
      for (int qf = 0; qf < 2; ++qf) {
        st[0][qf] = wmma16(ak0, bq[qf][ks], st[0][qf]);
        st[1][qf] = wmma16(ak1, bq[qf][ks], st[1][qf]);
      }
    }

    // Online softmax: per lane, column q = qf*16+lm; 16 keys in-lane + xor-16.
    float alpha[2];
#pragma unroll
    for (int qf = 0; qf < 2; ++qf) {
      float mx = st[0][qf][0];
#pragma unroll
      for (int kf = 0; kf < 2; ++kf)
#pragma unroll
        for (int r = 0; r < 8; ++r) mx = fmaxf(mx, st[kf][qf][r]);
      mx = fmaxf(mx, __shfl_xor(mx, 16, 32));
      float mnew = fmaxf(rm[qf], mx);
      alpha[qf] = __expf(rm[qf] - mnew);   // 0 on first tile
      rm[qf] = mnew;
      float ps = 0.0f;
#pragma unroll
      for (int kf = 0; kf < 2; ++kf)
#pragma unroll
        for (int r = 0; r < 8; ++r) {
          float p = __expf(st[kf][qf][r] - mnew);
          st[kf][qf][r] = p;
          ps += p;
        }
      ps += __shfl_xor(ps, 16, 32);
      rs[qf] = rs[qf] * alpha[qf] + ps;
    }

    // Rescale O: broadcast alpha from stats-lane (q&15) into O's C-layout.
#pragma unroll
    for (int f = 0; f < 2; ++f)
#pragma unroll
      for (int r = 0; r < 8; ++r) {
        float a = __shfl(alpha[f], r + 8 * lh, 32);
#pragma unroll
        for (int g = 0; g < 4; ++g) o[f][g][r] *= a;
      }

    // P^T (C-layout) -> LDS row-major P[q][key]; keys r..r+7 contiguous/lane.
#pragma unroll
    for (int kf = 0; kf < 2; ++kf)
#pragma unroll
      for (int qf = 0; qf < 2; ++qf) {
        v8h pk;
#pragma unroll
        for (int r = 0; r < 8; ++r) pk[r] = (half_t)st[kf][qf][r];
        *(v8h*)(pw + (qf * 16 + lm) * 32 + kf * 16 + 8 * lh) = pk;
      }
    __syncthreads();

    // O += P x V   (P A-frags from LDS, Vt rows contiguous in key axis)
    v16h pa0 = load_a(pw, 32, 0,  0, lane);
    v16h pa1 = load_a(pw, 32, 16, 0, lane);
#pragma unroll
    for (int g = 0; g < 4; ++g) {
      v16h bv = load_b(Vp, TOK, g * 16, kv, lane);
      o[0][g] = wmma16(pa0, bv, o[0][g]);
      o[1][g] = wmma16(pa1, bv, o[1][g]);
    }
    __syncthreads();
  }

  // Normalize by row sums (broadcast 1/l into O layout) and store token-major.
  float inv[2] = { 1.0f / rs[0], 1.0f / rs[1] };
#pragma unroll
  for (int f = 0; f < 2; ++f) {
#pragma unroll
    for (int r = 0; r < 8; ++r) {
      float iv = __shfl(inv[f], r + 8 * lh, 32);
      int m = q0 + f * 16 + lh * 8 + r;
      size_t base = (size_t)(b * SEQ + m) * HID + h * HDIM;
#pragma unroll
      for (int g = 0; g < 4; ++g)
        O[base + g * 16 + lm] = (half_t)(o[f][g][r] * iv);
    }
  }
}

// ---------------------------------------------------------------------------
// Launch
// ---------------------------------------------------------------------------
extern "C" void kernel_launch(void* const* d_in, const int* in_sizes, int n_in,
                              void* d_out, int out_size, void* d_ws, size_t ws_size,
                              hipStream_t stream) {
  (void)in_sizes; (void)n_in; (void)out_size; (void)ws_size;
  const float* x  = (const float*)d_in[0];
  const float* Wq = (const float*)d_in[1];
  const float* bq = (const float*)d_in[2];
  const float* Wk = (const float*)d_in[3];
  const float* bk = (const float*)d_in[4];
  const float* Wv = (const float*)d_in[5];
  const float* bv = (const float*)d_in[6];
  const float* Wo = (const float*)d_in[7];
  const float* bo = (const float*)d_in[8];

  const size_t SZ_X = (size_t)TOK * HID * sizeof(half_t);  // 12.6 MB
  const size_t SZ_W = (size_t)HID * HID * sizeof(half_t);  // 1.2 MB
  char* ws = (char*)d_ws;
  half_t* Xh  = (half_t*)ws; ws += SZ_X;
  half_t* Wqh = (half_t*)ws; ws += SZ_W;
  half_t* Wkh = (half_t*)ws; ws += SZ_W;
  half_t* Wvh = (half_t*)ws; ws += SZ_W;
  half_t* Woh = (half_t*)ws; ws += SZ_W;
  half_t* Qh  = (half_t*)ws; ws += SZ_X;
  half_t* Kh  = (half_t*)ws; ws += SZ_X;
  half_t* Vth = (half_t*)ws; ws += SZ_X;   // transposed [HID, TOK]
  half_t* Ah  = (half_t*)ws; ws += SZ_X;

  const int n4x = TOK * HID / 4;
  const int n4w = HID * HID / 4;
  cvt_f16<<<(n4x + 255) / 256, 256, 0, stream>>>(x,  Xh,  n4x);
  cvt_f16<<<(n4w + 255) / 256, 256, 0, stream>>>(Wq, Wqh, n4w);
  cvt_f16<<<(n4w + 255) / 256, 256, 0, stream>>>(Wk, Wkh, n4w);
  cvt_f16<<<(n4w + 255) / 256, 256, 0, stream>>>(Wv, Wvh, n4w);
  cvt_f16<<<(n4w + 255) / 256, 256, 0, stream>>>(Wo, Woh, n4w);

  dim3 gg(HID / 128, TOK / 128);
  gemm_wmma<0><<<gg, 256, 0, stream>>>(Xh, Wqh, bq, Qh,  TOK, HID, HID);
  gemm_wmma<0><<<gg, 256, 0, stream>>>(Xh, Wkh, bk, Kh,  TOK, HID, HID);
  gemm_wmma<1><<<gg, 256, 0, stream>>>(Xh, Wvh, bv, Vth, TOK, HID, HID);

  flash_attn<<<dim3(SEQ / 128, HEADS, BATCH), 128, 0, stream>>>(Qh, Kh, Vth, Ah);

  gemm_wmma<2><<<gg, 256, 0, stream>>>(Ah, Woh, bo, d_out, TOK, HID, HID);
}
